// TransformerBlock_66159676228162
// MI455X (gfx1250) — matrix operand order repaired and checked by reading, not compiled, not run on hardware
//
#include <hip/hip_runtime.h>
#include <hip/hip_bf16.h>
#include <math.h>

// ---------------------------------------------------------------------------
// Types for CDNA5 WMMA (wave32): 16x16x32 bf16 -> f32
// ---------------------------------------------------------------------------
typedef __attribute__((ext_vector_type(16))) __bf16        v16bf;
typedef __attribute__((ext_vector_type(8)))  float         v8f;
typedef __attribute__((ext_vector_type(4)))  unsigned int  u32x4;

union FragB16 {          // one WMMA A or B operand: 16 bf16 = 32 bytes
    v16bf v;
    u32x4 q[2];
};

__device__ __forceinline__ v8f wmma_bf16(v16bf a, v16bf b, v8f c) {
    // (neg_a, A, neg_b, B, c_mod, C, reuse_a, reuse_b)
    return __builtin_amdgcn_wmma_f32_16x16x32_bf16(false, a, false, b,
                                                   (short)0, c, false, false);
}

__device__ __forceinline__ __bf16 f2bf(float f) {
    unsigned u = __builtin_bit_cast(unsigned, f);
    unsigned r = u + 0x7fffu + ((u >> 16) & 1u);   // round-to-nearest-even
    unsigned short h = (unsigned short)(r >> 16);
    __bf16 b;
    __builtin_memcpy(&b, &h, sizeof(b));
    return b;
}

__device__ __forceinline__ float gelu_tanh(float x) {
    const float c = 0.7978845608028654f;           // sqrt(2/pi)
    return 0.5f * x * (1.0f + tanhf(c * (x + 0.044715f * x * x * x)));
}

__device__ __forceinline__ float waveSum16(float v) { // sum across 16-lane half
    v += __shfl_xor(v, 1, 32);
    v += __shfl_xor(v, 2, 32);
    v += __shfl_xor(v, 4, 32);
    v += __shfl_xor(v, 8, 32);
    return v;
}
__device__ __forceinline__ float waveMax16(float v) {
    v = fmaxf(v, __shfl_xor(v, 1, 32));
    v = fmaxf(v, __shfl_xor(v, 2, 32));
    v = fmaxf(v, __shfl_xor(v, 4, 32));
    v = fmaxf(v, __shfl_xor(v, 8, 32));
    return v;
}
__device__ __forceinline__ float waveSum32(float v) {
    v = waveSum16(v);
    v += __shfl_xor(v, 16, 32);
    return v;
}

// ---------------------------------------------------------------------------
// CDNA5 async global->LDS copy (ASYNCcnt-tracked; bypasses VGPR round-trip)
// ---------------------------------------------------------------------------
__device__ __forceinline__ void async_copy_b128(unsigned lds_off, const __bf16* gsrc) {
    asm volatile("global_load_async_to_lds_b128 %0, %1, off"
                 :: "v"(lds_off), "v"(gsrc) : "memory");
}
__device__ __forceinline__ void wait_async0() {
    asm volatile("s_wait_asynccnt 0x0" ::: "memory");
}

// ---------------------------------------------------------------------------
// fp32 -> bf16 weight conversion WITH transpose: W[K][N] -> Wt[N][K]
// (one-time, memory-bound; makes every GEMM B-fragment a contiguous load)
// ---------------------------------------------------------------------------
__global__ void cvt_t_kernel(const float* __restrict__ in, __bf16* __restrict__ out,
                             int K, int N) {
    int i = blockIdx.x * blockDim.x + threadIdx.x;
    int total = K * N;
    int stride = gridDim.x * blockDim.x;
    for (; i < total; i += stride) {
        int n = i / K, k = i - n * K;
        out[i] = f2bf(in[(size_t)k * N + n]);
    }
}

// ---------------------------------------------------------------------------
// LayerNorm over D=1024, one block (256 thr) per row, bf16 output
// ---------------------------------------------------------------------------
__global__ __launch_bounds__(256)
void ln_kernel(const float* __restrict__ x, const float* __restrict__ g,
               const float* __restrict__ sh, __bf16* __restrict__ out) {
    const int row = blockIdx.x;
    const int t = threadIdx.x;
    const float* xr = x + (size_t)row * 1024;
    float v0 = xr[t], v1 = xr[t + 256], v2 = xr[t + 512], v3 = xr[t + 768];
    float s  = v0 + v1 + v2 + v3;
    float ss = v0 * v0 + v1 * v1 + v2 * v2 + v3 * v3;
    s  = waveSum32(s);
    ss = waveSum32(ss);
    __shared__ float sb[8], ssb[8];
    if ((t & 31) == 0) { sb[t >> 5] = s; ssb[t >> 5] = ss; }
    __syncthreads();
    float S = 0.f, SS = 0.f;
#pragma unroll
    for (int i = 0; i < 8; ++i) { S += sb[i]; SS += ssb[i]; }
    float mean = S * (1.0f / 1024.0f);
    float var  = SS * (1.0f / 1024.0f) - mean * mean;
    float rstd = rsqrtf(var + 1e-5f);
    __bf16* orow = out + (size_t)row * 1024;
    orow[t]       = f2bf(g[t]       * (v0 - mean) * rstd + sh[t]);
    orow[t + 256] = f2bf(g[t + 256] * (v1 - mean) * rstd + sh[t + 256]);
    orow[t + 512] = f2bf(g[t + 512] * (v2 - mean) * rstd + sh[t + 512]);
    orow[t + 768] = f2bf(g[t + 768] * (v3 - mean) * rstd + sh[t + 768]);
}

// ---------------------------------------------------------------------------
// WMMA GEMM: C[M,N] = A[M,K] @ Bt[N,K]^T   (A row-major, B pre-transposed)
// 256 threads = 8 waves (2 M-groups x 4 N-groups); block tile 64(M) x 256(N).
// Wave tile: 32(M) x 64(N) -> 8 WMMAs/K-step from 2 A frags x 4 B frags
// (1.5 ds_load_b128 per WMMA). K-step 32.
// Staging: double-buffered LDS filled by global_load_async_to_lds_b128,
// overlapped with WMMA compute via ASYNCcnt.
// Epilogue: +bias, optional GELU, +residual; f32 / bf16 / transposed-bf16 out.
// ---------------------------------------------------------------------------
__global__ __launch_bounds__(256)
void gemm_bf16(const __bf16* __restrict__ A, const __bf16* __restrict__ Bt,
               const float* __restrict__ bias, const float* __restrict__ resid,
               float* __restrict__ outF, __bf16* __restrict__ outB,
               __bf16* __restrict__ outBT,
               int M, int N, int K, int gelu) {
    __shared__ __bf16 As[2][64][40];    // [buf][m][k], row padded to 80B
    __shared__ __bf16 Bs[2][256][40];   // [buf][n][k]

    const int tid  = threadIdx.x;
    const int lane = tid & 31;
    const int w    = tid >> 5;          // 0..7
    const int wm2  = w >> 2;            // M 32-group (0..1)
    const int wn4  = w & 3;             // N 64-group (0..3)
    const int half = lane >> 4;
    const int l16  = lane & 15;
    const int bm   = blockIdx.y * 64;
    const int bn   = blockIdx.x * 256;

    // staging map: thread t -> A chunk (sr, sc) + B chunks (sr + 64j, sc)
    const int sr = tid >> 2;            // 0..63
    const int sc = (tid & 3) * 8;       // 0,8,16,24

    const __bf16* aSrc = A + (size_t)(bm + sr) * K + sc;
    const __bf16* bSrc[4];
#pragma unroll
    for (int j = 0; j < 4; ++j)
        bSrc[j] = Bt + (size_t)(bn + sr + 64 * j) * K + sc;

    unsigned aDst[2], bDst[4][2];
#pragma unroll
    for (int bu = 0; bu < 2; ++bu) {
        aDst[bu] = (unsigned)(uintptr_t)&As[bu][sr][sc];
#pragma unroll
        for (int j = 0; j < 4; ++j)
            bDst[j][bu] = (unsigned)(uintptr_t)&Bs[bu][sr + 64 * j][sc];
    }

    auto stage = [&](int bu, int kk) {
        async_copy_b128(aDst[bu], aSrc + kk);
#pragma unroll
        for (int j = 0; j < 4; ++j)
            async_copy_b128(bDst[j][bu], bSrc[j] + kk);
    };

    v8f acc[2][4];
#pragma unroll
    for (int mt = 0; mt < 2; ++mt)
#pragma unroll
        for (int nt = 0; nt < 4; ++nt) acc[mt][nt] = v8f{};

    const int nk = K >> 5;
    stage(0, 0);
    for (int i = 0; i < nk; ++i) {
        wait_async0();                  // own wave's async writes to LDS done
        __syncthreads();                // all waves' writes visible
        if (i + 1 < nk) stage((i + 1) & 1, (i + 1) << 5);   // overlap next DMA

        const int bu = i & 1;
        FragB16 a[2];
#pragma unroll
        for (int mt = 0; mt < 2; ++mt) {
            const int mrow = wm2 * 32 + mt * 16 + l16;
            a[mt].q[0] = *(const u32x4*)&As[bu][mrow][half * 8];       // K 0..7/8..15
            a[mt].q[1] = *(const u32x4*)&As[bu][mrow][16 + half * 8];  // K 16../24..
        }
#pragma unroll
        for (int nt = 0; nt < 4; ++nt) {
            FragB16 b;
            const int nb = wn4 * 64 + nt * 16 + l16;
            b.q[0] = *(const u32x4*)&Bs[bu][nb][half * 8];             // K 0..7/8..15
            b.q[1] = *(const u32x4*)&Bs[bu][nb][16 + half * 8];
            acc[0][nt] = wmma_bf16(a[0].v, b.v, acc[0][nt]);
            acc[1][nt] = wmma_bf16(a[1].v, b.v, acc[1][nt]);
        }
        __syncthreads();                // done reading buf before reuse
    }

#pragma unroll
    for (int mt = 0; mt < 2; ++mt) {
#pragma unroll
        for (int nt = 0; nt < 4; ++nt) {
#pragma unroll
            for (int r = 0; r < 8; ++r) {
                const size_t row = (size_t)(bm + wm2 * 32 + mt * 16 + r + half * 8);
                const size_t col = (size_t)(bn + wn4 * 64 + nt * 16 + l16);
                float v = acc[mt][nt][r];
                if (bias)  v += bias[col];
                if (gelu)  v = gelu_tanh(v);
                if (resid) v += resid[row * N + col];
                if (outF)  outF[row * N + col] = v;
                if (outB)  outB[row * N + col] = f2bf(v);
                if (outBT) outBT[col * M + row] = f2bf(v);   // transposed (Vt)
            }
        }
    }
}

// ---------------------------------------------------------------------------
// Causal flash attention: S=2048, H=16, HD=64, scale=1/8.
// 128 threads = 4 waves; each wave owns one 16-query tile; 32-key steps.
// Q.K^T and P.V via v_wmma_f32_16x16x32_bf16. V is pre-transposed (Vt[d][s])
// so P.V B-fragments are contiguous 16B loads. P converted C-layout->A-layout
// through wave-private LDS (in-order DS, s_wait_dscnt).
// ---------------------------------------------------------------------------
__global__ __launch_bounds__(128)
void attn_kernel(const __bf16* __restrict__ Q, const __bf16* __restrict__ Km,
                 const __bf16* __restrict__ Vt, __bf16* __restrict__ O) {
    __shared__ __bf16 P[4][16][40];     // per-wave probability tile

    const int tid  = threadIdx.x;
    const int lane = tid & 31;
    const int w    = tid >> 5;
    const int half = lane >> 4;
    const int l16  = lane & 15;
    const int b    = blockIdx.z;
    const int h    = blockIdx.y;
    const int q0   = (blockIdx.x * 4 + w) * 16;
    const float scale = 0.125f;         // 1/sqrt(64)

    const size_t base = (size_t)b * 2048 * 1024 + (size_t)h * 64;

    // Q fragments (A-layout, M = l16): d-steps 0..31 and 32..63
    FragB16 aq0, aq1;
    {
        const __bf16* qrow = Q + base + (size_t)(q0 + l16) * 1024;
        aq0.q[0] = *(const u32x4*)(qrow + half * 8);
        aq0.q[1] = *(const u32x4*)(qrow + 16 + half * 8);
        aq1.q[0] = *(const u32x4*)(qrow + 32 + half * 8);
        aq1.q[1] = *(const u32x4*)(qrow + 48 + half * 8);
    }

    v8f o0 = {}, o1 = {}, o2 = {}, o3 = {};
    float mrun[8], lrun[8];
#pragma unroll
    for (int r = 0; r < 8; ++r) { mrun[r] = -1e30f; lrun[r] = 0.f; }

    const int kend = q0 + 16;
    for (int k0 = 0; k0 < kend; k0 += 32) {
        // ---- scores: Q(16x64) @ K^T(64x32) -> two 16x16 C tiles ----
        v8f s0 = {}, s1 = {};
        {
            FragB16 bk;
            const __bf16* kr0 = Km + base + (size_t)(k0 + l16) * 1024;
            const __bf16* kr1 = Km + base + (size_t)(k0 + 16 + l16) * 1024;
            bk.q[0] = *(const u32x4*)(kr0 + half * 8);
            bk.q[1] = *(const u32x4*)(kr0 + 16 + half * 8);
            s0 = wmma_bf16(aq0.v, bk.v, s0);
            bk.q[0] = *(const u32x4*)(kr0 + 32 + half * 8);
            bk.q[1] = *(const u32x4*)(kr0 + 48 + half * 8);
            s0 = wmma_bf16(aq1.v, bk.v, s0);
            bk.q[0] = *(const u32x4*)(kr1 + half * 8);
            bk.q[1] = *(const u32x4*)(kr1 + 16 + half * 8);
            s1 = wmma_bf16(aq0.v, bk.v, s1);
            bk.q[0] = *(const u32x4*)(kr1 + 32 + half * 8);
            bk.q[1] = *(const u32x4*)(kr1 + 48 + half * 8);
            s1 = wmma_bf16(aq1.v, bk.v, s1);
        }

        // ---- online softmax (row = VGPR index r within lane half) ----
#pragma unroll
        for (int r = 0; r < 8; ++r) {
            const int qrow = q0 + r + half * 8;
            const int key0 = k0 + l16;
            const int key1 = k0 + 16 + l16;
            float v0 = (key0 <= qrow) ? s0[r] * scale : -1e30f;
            float v1 = (key1 <= qrow) ? s1[r] * scale : -1e30f;
            float mx = waveMax16(fmaxf(v0, v1));
            float mnew  = fmaxf(mrun[r], mx);
            float alpha = __expf(mrun[r] - mnew);
            float p0 = __expf(v0 - mnew);
            float p1 = __expf(v1 - mnew);
            float ps = waveSum16(p0 + p1);
            lrun[r] = lrun[r] * alpha + ps;
            mrun[r] = mnew;
            o0[r] *= alpha; o1[r] *= alpha; o2[r] *= alpha; o3[r] *= alpha;
            P[w][r + half * 8][l16]      = f2bf(p0);
            P[w][r + half * 8][16 + l16] = f2bf(p1);
        }
        asm volatile("s_wait_dscnt 0x0" ::: "memory");  // wave-private LDS RAW

        // ---- P (A-layout from LDS) @ V(32x64) via Vt[d][s] ----
        FragB16 ap;
        ap.q[0] = *(const u32x4*)&P[w][l16][half * 8];
        ap.q[1] = *(const u32x4*)&P[w][l16][16 + half * 8];

        // B-frag: b[j] = V[key=k0+half*8+j (+8 for j>=8)][d] = Vt[d][b*2048+key]
        const __bf16* vbase = Vt + (size_t)(h * 64 + l16) * 4096
                                 + (size_t)b * 2048 + k0 + half * 8;
        FragB16 bv;
        bv.q[0] = *(const u32x4*)(vbase);
        bv.q[1] = *(const u32x4*)(vbase + 16);
        o0 = wmma_bf16(ap.v, bv.v, o0);
        bv.q[0] = *(const u32x4*)(vbase + (size_t)16 * 4096);
        bv.q[1] = *(const u32x4*)(vbase + (size_t)16 * 4096 + 16);
        o1 = wmma_bf16(ap.v, bv.v, o1);
        bv.q[0] = *(const u32x4*)(vbase + (size_t)32 * 4096);
        bv.q[1] = *(const u32x4*)(vbase + (size_t)32 * 4096 + 16);
        o2 = wmma_bf16(ap.v, bv.v, o2);
        bv.q[0] = *(const u32x4*)(vbase + (size_t)48 * 4096);
        bv.q[1] = *(const u32x4*)(vbase + (size_t)48 * 4096 + 16);
        o3 = wmma_bf16(ap.v, bv.v, o3);
    }

    // ---- normalize and store ctx (bf16, row-major [s][d]) ----
#pragma unroll
    for (int r = 0; r < 8; ++r) {
        const float inv = 1.0f / lrun[r];
        const size_t row = (size_t)b * 2048 + q0 + r + half * 8;
        const size_t col = (size_t)h * 64 + l16;
        __bf16* orow = O + row * 1024 + col;
        orow[0]  = f2bf(o0[r] * inv);
        orow[16] = f2bf(o1[r] * inv);
        orow[32] = f2bf(o2[r] * inv);
        orow[48] = f2bf(o3[r] * inv);
    }
}

// ---------------------------------------------------------------------------
// Host-side orchestration
// ---------------------------------------------------------------------------
extern "C" void kernel_launch(void* const* d_in, const int* in_sizes, int n_in,
                              void* d_out, int out_size, void* d_ws, size_t ws_size,
                              hipStream_t stream) {
    const float* x  = (const float*)d_in[0];
    const float* Wq = (const float*)d_in[1];
    const float* Wk = (const float*)d_in[2];
    const float* Wv = (const float*)d_in[3];
    const float* Wo = (const float*)d_in[4];
    const float* bo = (const float*)d_in[5];
    const float* W1 = (const float*)d_in[6];
    const float* b1 = (const float*)d_in[7];
    const float* W2 = (const float*)d_in[8];
    const float* b2 = (const float*)d_in[9];
    const float* g1 = (const float*)d_in[10];
    const float* s1 = (const float*)d_in[11];
    const float* g2 = (const float*)d_in[12];
    const float* s2 = (const float*)d_in[13];

    const int D = 1024, F = 4096, ROWS = 2 * 2048;
    char* ws = (char*)d_ws;
    size_t off = 0;
    auto alloc = [&](size_t bytes) -> char* {
        char* p = ws + off;
        off += (bytes + 255) & ~(size_t)255;
        return p;
    };
    __bf16* wqt  = (__bf16*)alloc((size_t)D * D * 2);   // transposed bf16 weights
    __bf16* wkt  = (__bf16*)alloc((size_t)D * D * 2);
    __bf16* wvt  = (__bf16*)alloc((size_t)D * D * 2);
    __bf16* wot  = (__bf16*)alloc((size_t)D * D * 2);
    __bf16* w1t  = (__bf16*)alloc((size_t)D * F * 2);
    __bf16* w2t  = (__bf16*)alloc((size_t)F * D * 2);
    __bf16* hb   = (__bf16*)alloc((size_t)ROWS * D * 2);
    __bf16* qb   = (__bf16*)alloc((size_t)ROWS * D * 2);
    __bf16* kb   = (__bf16*)alloc((size_t)ROWS * D * 2);
    __bf16* vt   = (__bf16*)alloc((size_t)D * ROWS * 2); // Vt[d][b*2048+s]
    __bf16* ctxb = (__bf16*)alloc((size_t)ROWS * D * 2);
    float*  x1   = (float*) alloc((size_t)ROWS * D * 4);
    __bf16* h2b  = (__bf16*)alloc((size_t)ROWS * D * 2);
    __bf16* ffb  = (__bf16*)alloc((size_t)ROWS * F * 2);

    // 1) weights -> bf16, transposed to [N][K]
    cvt_t_kernel<<<1024, 256, 0, stream>>>(Wq, wqt, D, D);
    cvt_t_kernel<<<1024, 256, 0, stream>>>(Wk, wkt, D, D);
    cvt_t_kernel<<<1024, 256, 0, stream>>>(Wv, wvt, D, D);
    cvt_t_kernel<<<1024, 256, 0, stream>>>(Wo, wot, D, D);
    cvt_t_kernel<<<2048, 256, 0, stream>>>(W1, w1t, D, F);
    cvt_t_kernel<<<2048, 256, 0, stream>>>(W2, w2t, F, D);

    // 2) LN1
    ln_kernel<<<ROWS, 256, 0, stream>>>(x, g1, s1, hb);

    // 3) QKV projections (V written transposed for attention)
    dim3 gq(D / 256, ROWS / 64);
    gemm_bf16<<<gq, 256, 0, stream>>>(hb, wqt, nullptr, nullptr, nullptr, qb, nullptr, ROWS, D, D, 0);
    gemm_bf16<<<gq, 256, 0, stream>>>(hb, wkt, nullptr, nullptr, nullptr, kb, nullptr, ROWS, D, D, 0);
    gemm_bf16<<<gq, 256, 0, stream>>>(hb, wvt, nullptr, nullptr, nullptr, nullptr, vt, ROWS, D, D, 0);

    // 4) causal flash attention
    dim3 ga(2048 / 64, 16, 2);
    attn_kernel<<<ga, 128, 0, stream>>>(qb, kb, vt, ctxb);

    // 5) output projection + bias + residual -> x1 (f32)
    gemm_bf16<<<gq, 256, 0, stream>>>(ctxb, wot, bo, x, x1, nullptr, nullptr, ROWS, D, D, 0);

    // 6) LN2
    ln_kernel<<<ROWS, 256, 0, stream>>>(x1, g2, s2, h2b);

    // 7) FFN up + GELU (bf16 out)
    dim3 g1d(F / 256, ROWS / 64);
    gemm_bf16<<<g1d, 256, 0, stream>>>(h2b, w1t, b1, nullptr, nullptr, ffb, nullptr, ROWS, F, D, 1);

    // 8) FFN down + bias + residual -> d_out (f32)
    gemm_bf16<<<gq, 256, 0, stream>>>(ffb, w2t, b2, x1, (float*)d_out, nullptr, nullptr, ROWS, D, F, 0);
}
